// Model_37529424233188
// MI455X (gfx1250) — compile-verified, matrix operand order checked by reading, not run on hardware
//
#include <hip/hip_runtime.h>
#include <math.h>

#define H_    4
#define DIM_  128
#define DH_   32
#define L_    1024
#define N_    64
#define B_    4
#define C_    16
#define T_    1024
#define DFF_  256
#define PRED_ 96
#define KTOP_ 102   // int(0.1 * 1024)

typedef __attribute__((ext_vector_type(16))) __bf16 v16bf;
typedef __attribute__((ext_vector_type(8)))  __bf16 v8bf;
typedef __attribute__((ext_vector_type(8)))  float  v8f;

// ---- WMMA 16x16x32 bf16 tile load (wave32) ----
// Per-lane A/B layout: row = lane&15 (M for A, N for pre-transposed B),
// elements e=0..7 -> K = half*8 + e ; e=8..15 -> K = 16 + half*8 + (e-8).
// With bf16 row-major source these are two contiguous 16-byte runs -> 2x B128 loads.
__device__ __forceinline__ v16bf load_tile16(const __bf16* src, int ld, int lane) {
    int row = lane & 15, half = lane >> 4;
    const __bf16* base = src + (size_t)row * ld + half * 8;
    v8bf lo = *(const v8bf*)(base);
    v8bf hi = *(const v8bf*)(base + 16);
    v16bf t;
#pragma unroll
    for (int e = 0; e < 8; ++e) { t[e] = lo[e]; t[e + 8] = hi[e]; }
    return t;
}

// K-loop GEMM core: A tile rows at `a` (lda), B tile cols at `bt` (pre-transposed, ldb)
__device__ __forceinline__ v8f wmma_kloop(const __bf16* a, int lda, const __bf16* bt, int ldb,
                                          int Kt, int lane) {
    v8f acc = {};
    for (int k0 = 0; k0 < Kt; k0 += 32) {
        v16bf Av = load_tile16(a + k0, lda, lane);
        v16bf Bv = load_tile16(bt + k0, ldb, lane);
        acc = __builtin_amdgcn_wmma_f32_16x16x32_bf16(false, Av, false, Bv, (short)0, acc, false, false);
    }
    return acc;
}

__device__ __forceinline__ float gelu_exact(float x) {
    return 0.5f * x * (1.0f + erff(x * 0.7071067811865475f));
}

__device__ __forceinline__ float block_reduce128(float v, float* sred, int tid) {
    sred[tid] = v; __syncthreads();
    for (int s = 64; s > 0; s >>= 1) {
        if (tid < s) sred[tid] += sred[tid + s];
        __syncthreads();
    }
    float r = sred[0]; __syncthreads();
    return r;
}

// ---------------- RevIN stats ----------------
__global__ void revin_stats_kernel(const float* __restrict__ x, float* __restrict__ meanb,
                                   float* __restrict__ stdb) {
    __shared__ float s1[256], s2[256];
    int bc = blockIdx.x, b = bc / C_, c = bc % C_;
    const float* xp = x + (size_t)b * T_ * C_ + c;
    float a = 0.f, a2 = 0.f;
    for (int t = threadIdx.x; t < T_; t += 256) {
        float v = xp[(size_t)t * C_];
        a += v; a2 += v * v;
    }
    s1[threadIdx.x] = a; s2[threadIdx.x] = a2; __syncthreads();
    for (int s = 128; s > 0; s >>= 1) {
        if (threadIdx.x < s) { s1[threadIdx.x] += s1[threadIdx.x + s]; s2[threadIdx.x] += s2[threadIdx.x + s]; }
        __syncthreads();
    }
    if (threadIdx.x == 0) {
        float m = s1[0] * (1.f / T_);
        float var = s2[0] * (1.f / T_) - m * m;
        meanb[bc] = m;
        stdb[bc] = sqrtf(var + 1e-5f);
    }
}

// ---------------- patchify + embed + PE ----------------
__global__ void patch_embed_kernel(const float* __restrict__ x, const float* __restrict__ meanb,
                                   const float* __restrict__ stdb, const float* __restrict__ pw,
                                   const float* __restrict__ pb, float* __restrict__ h) {
    int bl = blockIdx.x, b = bl >> 10, l = bl & (L_ - 1);
    int d = threadIdx.x;
    int c = l / N_;
    int t0 = (l % N_) * 16;
    float m = meanb[b * C_ + c], sd = stdb[b * C_ + c];
    float acc = pb[d];
#pragma unroll
    for (int p = 0; p < 16; ++p) {
        float xv = (x[((size_t)b * T_ + t0 + p) * C_ + c] - m) / sd;
        acc += xv * pw[p * DIM_ + d];
    }
    int i2 = d & ~1;
    float div = expf((float)i2 * (-9.210340371976184f / (float)DIM_));
    float ang = (float)l * div;
    float pe = (d & 1) ? cosf(ang) : sinf(ang);
    h[(size_t)bl * DIM_ + d] = acc + pe;
}

// ---------------- LN1 + q/k/vT projections; bf16 outputs ----------------
__global__ void ln_qkv_kernel(const float* __restrict__ h, const float* __restrict__ g,
                              const float* __restrict__ bta, const float* __restrict__ gw,
                              const float* __restrict__ gb, const float* __restrict__ p1w,
                              const float* __restrict__ p1b, const float* __restrict__ p2w,
                              const float* __restrict__ p2b, __bf16* __restrict__ qbf,
                              __bf16* __restrict__ kbf, __bf16* __restrict__ vTbf) {
    __shared__ float sy[DIM_];
    __shared__ float sred[DIM_];
    int bl = blockIdx.x, d = threadIdx.x;
    float val = h[(size_t)bl * DIM_ + d];
    float mu = block_reduce128(val, sred, d) * (1.f / DIM_);
    float cdev = val - mu;
    float var = block_reduce128(cdev * cdev, sred, d) * (1.f / DIM_);
    float yv = cdev * rsqrtf(var + 1e-5f) * g[d] + bta[d];
    sy[d] = yv; __syncthreads();
    int b = bl >> 10, l = bl & (L_ - 1);
    // v = y @ gcn_w + gcn_b, stored TRANSPOSED per batch: vT[b][d][l]
    float acc = gb[d];
    for (int e = 0; e < DIM_; ++e) acc += sy[e] * gw[e * DIM_ + d];
    vTbf[((size_t)b * DIM_ + d) * L_ + l] = (__bf16)acc;
    // per-head q,k
    int hd = d >> 5, j = d & 31;
    float qa = p1b[j], ka = p2b[j];
    const float* yh = sy + hd * DH_;
    for (int m2 = 0; m2 < DH_; ++m2) {
        float ym = yh[m2];
        qa += ym * p1w[m2 * DH_ + j];
        ka += ym * p2w[m2 * DH_ + j];
    }
    size_t qi = (((size_t)(b * H_ + hd)) * L_ + l) * DH_ + j;
    qbf[qi] = (__bf16)qa;
    kbf[qi] = (__bf16)ka;
}

// ---------------- LN2 -> bf16 ----------------
__global__ void ln_bf16_kernel(const float* __restrict__ h, const float* __restrict__ g,
                               const float* __restrict__ bta, __bf16* __restrict__ ybf) {
    __shared__ float sred[DIM_];
    int bl = blockIdx.x, d = threadIdx.x;
    float val = h[(size_t)bl * DIM_ + d];
    float mu = block_reduce128(val, sred, d) * (1.f / DIM_);
    float cdev = val - mu;
    float var = block_reduce128(cdev * cdev, sred, d) * (1.f / DIM_);
    ybf[(size_t)bl * DIM_ + d] = (__bf16)(cdev * rsqrtf(var + 1e-5f) * g[d] + bta[d]);
}

// ---------------- adj = gelu(q @ k^T): WMMA, A reused across 4 N-tiles ----------------
__global__ void attn_scores_kernel(const __bf16* __restrict__ qbf, const __bf16* __restrict__ kbf,
                                   float* __restrict__ adjf) {
    int lane = threadIdx.x;
    int tm = blockIdx.x, tn4 = blockIdx.y, bh = blockIdx.z;
    v16bf Av = load_tile16(qbf + ((size_t)bh * L_ + tm * 16) * DH_, DH_, lane);
    float* ap = adjf + (size_t)bh * L_ * L_;
#pragma unroll
    for (int c4 = 0; c4 < 4; ++c4) {
        int tn = tn4 * 4 + c4;
        v16bf Bv = load_tile16(kbf + ((size_t)bh * L_ + tn * 16) * DH_, DH_, lane);
        v8f acc = {};
        acc = __builtin_amdgcn_wmma_f32_16x16x32_bf16(false, Av, false, Bv, (short)0, acc, false, false);
        int col = tn * 16 + (lane & 15);
#pragma unroll
        for (int r = 0; r < 8; ++r) {
            int row = tm * 16 + ((lane < 16) ? r : r + 8);
            ap[(size_t)row * L_ + col] = gelu_exact(acc[r]);
        }
    }
}

// ---- fused per-row: top-k threshold, top-p expert gating, structural mask, softmax.
// ---- (L1-normalize after softmax is an identity.) Emits bf16 for the adj@v WMMA. ----
__global__ void topk_gate_softmax_kernel(const float* __restrict__ adjf, __bf16* __restrict__ adjbf,
                                         const float* __restrict__ gw, const float* __restrict__ masks) {
    __shared__ float srow[L_];
    __shared__ float ssort[L_];
    __shared__ float sr0[256], sr1[256], sr2[256];
    __shared__ float sgate[4];
    int tid = threadIdx.x;
    int rowid = blockIdx.x;
    int l = rowid & (L_ - 1);
    const float* arow = adjf + (size_t)rowid * L_;
    for (int j = tid; j < L_; j += 256) { float a = arow[j]; srow[j] = a; ssort[j] = a; }
    __syncthreads();
    // bitonic sort ascending
    for (int k2 = 2; k2 <= L_; k2 <<= 1) {
        for (int jj = k2 >> 1; jj > 0; jj >>= 1) {
            for (int idx = tid; idx < L_; idx += 256) {
                int ixj = idx ^ jj;
                if (ixj > idx) {
                    bool up = ((idx & k2) == 0);
                    float a = ssort[idx], b = ssort[ixj];
                    if ((a > b) == up) { ssort[idx] = b; ssort[ixj] = a; }
                }
            }
            __syncthreads();
        }
    }
    float thresh = ssort[KTOP_ - 1];
    float g0 = 0.f, g1 = 0.f, g2 = 0.f;
    for (int j = tid; j < L_; j += 256) {
        float a = srow[j];
        a = (a > thresh) ? a : 0.f;
        srow[j] = a;
        const float* gwj = gw + j * 3;
        g0 += a * gwj[0]; g1 += a * gwj[1]; g2 += a * gwj[2];
    }
    sr0[tid] = g0; sr1[tid] = g1; sr2[tid] = g2; __syncthreads();
    for (int s = 128; s > 0; s >>= 1) {
        if (tid < s) { sr0[tid] += sr0[tid + s]; sr1[tid] += sr1[tid + s]; sr2[tid] += sr2[tid + s]; }
        __syncthreads();
    }
    if (tid == 0) {
        float p[3];
        float mx = fmaxf(sr0[0], fmaxf(sr1[0], sr2[0]));
        p[0] = __expf(sr0[0] - mx); p[1] = __expf(sr1[0] - mx); p[2] = __expf(sr2[0] - mx);
        float s = p[0] + p[1] + p[2];
        p[0] /= s; p[1] /= s; p[2] /= s;
        int ord[3] = {0, 1, 2};
        if (p[ord[0]] < p[ord[1]]) { int t = ord[0]; ord[0] = ord[1]; ord[1] = t; }
        if (p[ord[1]] < p[ord[2]]) { int t = ord[1]; ord[1] = ord[2]; ord[2] = t; }
        if (p[ord[0]] < p[ord[1]]) { int t = ord[0]; ord[0] = ord[1]; ord[1] = t; }
        float cum = 0.f; bool m[3];
        for (int t2 = 0; t2 < 3; ++t2) { cum += p[ord[t2]]; m[t2] = cum > 0.5f; }
        int thr = 0;
        for (int t2 = 0; t2 < 3; ++t2) { if (m[t2]) { thr = t2; break; } }
        m[thr] = false;
        float gts[3] = {0.f, 0.f, 0.f};
        for (int t2 = 0; t2 < 3; ++t2) if (!m[t2]) gts[ord[t2]] = 1.f;
        sgate[0] = gts[0]; sgate[1] = gts[1]; sgate[2] = gts[2];
    }
    __syncthreads();
    float ga = sgate[0], gb2 = sgate[1], gc = sgate[2];
    const float* mr = masks + (size_t)l * 3 * L_;
    float lmax = -3.4e38f;
    for (int j = tid; j < L_; j += 256) {
        float mm = ga * mr[j] + gb2 * mr[L_ + j] + gc * mr[2 * L_ + j] + ((j == l) ? 1.f : 0.f);
        float a = srow[j] * mm;
        srow[j] = a;
        lmax = fmaxf(lmax, a);
    }
    sr0[tid] = lmax; __syncthreads();
    for (int s = 128; s > 0; s >>= 1) { if (tid < s) sr0[tid] = fmaxf(sr0[tid], sr0[tid + s]); __syncthreads(); }
    float rmax = sr0[0]; __syncthreads();
    float lsum = 0.f;
    for (int j = tid; j < L_; j += 256) { float e = __expf(srow[j] - rmax); srow[j] = e; lsum += e; }
    sr0[tid] = lsum; __syncthreads();
    for (int s = 128; s > 0; s >>= 1) { if (tid < s) sr0[tid] += sr0[tid + s]; __syncthreads(); }
    float inv = 1.f / sr0[0];
    __bf16* brow = adjbf + (size_t)rowid * L_;
    for (int j = tid; j < L_; j += 256) brow[j] = (__bf16)(srow[j] * inv);
}

// ---------------- h += adj @ v (per head): WMMA, B128 loads only ----------------
__global__ void attn_apply_kernel(const __bf16* __restrict__ adjbf, const __bf16* __restrict__ vTbf,
                                  float* __restrict__ h) {
    int lane = threadIdx.x;
    int tm = blockIdx.x, bh = blockIdx.y;
    int b = bh / H_, hd = bh % H_;
    const __bf16* ap = adjbf + (size_t)bh * L_ * L_ + (size_t)(tm * 16) * L_;
    const __bf16* v0 = vTbf + ((size_t)b * DIM_ + hd * DH_) * L_;
    v8f acc0 = {}, acc1 = {};
    for (int k0 = 0; k0 < L_; k0 += 32) {
        v16bf Av = load_tile16(ap + k0, L_, lane);
        v16bf B0 = load_tile16(v0 + k0, L_, lane);
        v16bf B1 = load_tile16(v0 + (size_t)16 * L_ + k0, L_, lane);
        acc0 = __builtin_amdgcn_wmma_f32_16x16x32_bf16(false, Av, false, B0, (short)0, acc0, false, false);
        acc1 = __builtin_amdgcn_wmma_f32_16x16x32_bf16(false, Av, false, B1, (short)0, acc1, false, false);
    }
    int col = lane & 15;
#pragma unroll
    for (int r = 0; r < 8; ++r) {
        int row = tm * 16 + ((lane < 16) ? r : r + 8);
        float* hp = h + ((size_t)b * L_ + row) * DIM_ + hd * DH_;
        hp[col] += acc0[r];
        hp[16 + col] += acc1[r];
    }
}

// ---------------- FFN1: mid = gelu(y @ W1 + b1) -> bf16 ----------------
__global__ void ffn1_wmma_kernel(const __bf16* __restrict__ ybf, const __bf16* __restrict__ w1T,
                                 const float* __restrict__ bias, __bf16* __restrict__ midbf) {
    int lane = threadIdx.x;
    int tm = blockIdx.x, tn = blockIdx.y;
    v8f acc = wmma_kloop(ybf + (size_t)(tm * 16) * DIM_, DIM_,
                         w1T + (size_t)(tn * 16) * DIM_, DIM_, DIM_, lane);
    int col = lane & 15;
    int cidx = tn * 16 + col;
    float bv = bias[cidx];
#pragma unroll
    for (int r = 0; r < 8; ++r) {
        int row = tm * 16 + ((lane < 16) ? r : r + 8);
        midbf[(size_t)row * DFF_ + cidx] = (__bf16)gelu_exact(acc[r] + bv);
    }
}

// ---------------- FFN2: h += mid @ W2 + b2 ----------------
__global__ void ffn2_wmma_kernel(const __bf16* __restrict__ midbf, const __bf16* __restrict__ w2T,
                                 const float* __restrict__ bias, float* __restrict__ h) {
    int lane = threadIdx.x;
    int tm = blockIdx.x, tn = blockIdx.y;
    v8f acc = wmma_kloop(midbf + (size_t)(tm * 16) * DFF_, DFF_,
                         w2T + (size_t)(tn * 16) * DFF_, DFF_, DFF_, lane);
    int col = lane & 15;
    int cidx = tn * 16 + col;
    float bv = bias[cidx];
#pragma unroll
    for (int r = 0; r < 8; ++r) {
        int row = tm * 16 + ((lane < 16) ? r : r + 8);
        h[(size_t)row * DIM_ + cidx] += acc[r] + bv;
    }
}

// ---------------- head (WMMA) + RevIN denorm fused into store ----------------
// A rows (b*16 + c) are contiguous 8192-length slices of h; that row index
// equals the RevIN (b,c) stats index.
__global__ void head_wmma_kernel(const __bf16* __restrict__ hbf, const __bf16* __restrict__ hwT,
                                 const float* __restrict__ hb, const float* __restrict__ meanb,
                                 const float* __restrict__ stdb, float* __restrict__ out) {
    int lane = threadIdx.x;
    int tm = blockIdx.x, tn = blockIdx.y;
    const int KH = N_ * DIM_; // 8192
    v8f acc = wmma_kloop(hbf + (size_t)(tm * 16) * KH, KH,
                         hwT + (size_t)(tn * 16) * KH, KH, KH, lane);
    int col = lane & 15;
    int p = tn * 16 + col;
    float bv = hb[p];
#pragma unroll
    for (int r = 0; r < 8; ++r) {
        int rb = tm * 16 + ((lane < 16) ? r : r + 8);
        int b = rb >> 4, c = rb & 15;
        out[((size_t)b * PRED_ + p) * C_ + c] = (acc[r] + bv) * stdb[rb] + meanb[rb];
    }
}

// ---------------- utility: f32 [K,N] -> bf16 transposed [N,K] ----------------
__global__ void transpose_bf16_kernel(const float* __restrict__ w, __bf16* __restrict__ wt,
                                      int K, int Nn) {
    int idx = blockIdx.x * 256 + threadIdx.x;
    if (idx < K * Nn) {
        int k = idx / Nn, n = idx % Nn;
        wt[(size_t)n * K + k] = (__bf16)w[idx];
    }
}

__global__ void cast_bf16_kernel(const float* __restrict__ src, __bf16* __restrict__ dst, int n) {
    int idx = blockIdx.x * 256 + threadIdx.x;
    if (idx < n) dst[idx] = (__bf16)src[idx];
}

extern "C" void kernel_launch(void* const* d_in, const int* in_sizes, int n_in,
                              void* d_out, int out_size, void* d_ws, size_t ws_size,
                              hipStream_t stream) {
    (void)in_sizes; (void)n_in; (void)out_size; (void)ws_size;
    const float* x       = (const float*)d_in[0];
    const float* masks   = (const float*)d_in[1];
    const float* patch_w = (const float*)d_in[2];
    const float* patch_b = (const float*)d_in[3];
    const float* p1_w    = (const float*)d_in[4];
    const float* p1_b    = (const float*)d_in[5];
    const float* p2_w    = (const float*)d_in[6];
    const float* p2_b    = (const float*)d_in[7];
    const float* gate_w  = (const float*)d_in[8];
    const float* gcn_w   = (const float*)d_in[9];
    const float* gcn_b   = (const float*)d_in[10];
    const float* ln1_g   = (const float*)d_in[11];
    const float* ln1_b   = (const float*)d_in[12];
    const float* ln2_g   = (const float*)d_in[13];
    const float* ln2_b   = (const float*)d_in[14];
    const float* ffn1_w  = (const float*)d_in[15];
    const float* ffn1_b  = (const float*)d_in[16];
    const float* ffn2_w  = (const float*)d_in[17];
    const float* ffn2_b  = (const float*)d_in[18];
    const float* head_w  = (const float*)d_in[19];
    const float* head_b  = (const float*)d_in[20];
    float* out = (float*)d_out;

    // ---- workspace layout (f32 region first, then bf16 region) ----
    float* wsf = (float*)d_ws;
    float* meanb = wsf;  wsf += 64;
    float* stdb  = wsf;  wsf += 64;
    float* hbuf  = wsf;  wsf += (size_t)B_ * L_ * DIM_;        // 512K f32
    float* adjf  = wsf;  wsf += (size_t)B_ * H_ * L_ * L_;     // 16M f32
    __bf16* wsb = (__bf16*)wsf;
    __bf16* qbf   = wsb;  wsb += (size_t)B_ * H_ * L_ * DH_;   // 512K
    __bf16* kbf   = wsb;  wsb += (size_t)B_ * H_ * L_ * DH_;   // 512K
    __bf16* vTbf  = wsb;  wsb += (size_t)B_ * DIM_ * L_;       // 512K
    __bf16* ybf   = wsb;  wsb += (size_t)B_ * L_ * DIM_;       // 512K
    __bf16* midbf = wsb;  wsb += (size_t)B_ * L_ * DFF_;       // 1M
    __bf16* hbf   = wsb;  wsb += (size_t)B_ * L_ * DIM_;       // 512K
    __bf16* w1T   = wsb;  wsb += (size_t)DIM_ * DFF_;          // 32K
    __bf16* w2T   = wsb;  wsb += (size_t)DFF_ * DIM_;          // 32K
    __bf16* hwT   = wsb;  wsb += (size_t)N_ * DIM_ * PRED_;    // 768K
    __bf16* adjbf = wsb;                                       // 16M

    revin_stats_kernel<<<B_ * C_, 256, 0, stream>>>(x, meanb, stdb);
    patch_embed_kernel<<<B_ * L_, DIM_, 0, stream>>>(x, meanb, stdb, patch_w, patch_b, hbuf);

    for (int i = 0; i < 3; ++i) {
        ln_qkv_kernel<<<B_ * L_, DIM_, 0, stream>>>(
            hbuf, ln1_g + i * DIM_, ln1_b + i * DIM_,
            gcn_w + (size_t)i * DIM_ * DIM_, gcn_b + i * DIM_,
            p1_w + i * DH_ * DH_, p1_b + i * DH_,
            p2_w + i * DH_ * DH_, p2_b + i * DH_,
            qbf, kbf, vTbf);
        attn_scores_kernel<<<dim3(L_ / 16, L_ / 64, B_ * H_), 32, 0, stream>>>(qbf, kbf, adjf);
        topk_gate_softmax_kernel<<<B_ * H_ * L_, 256, 0, stream>>>(
            adjf, adjbf, gate_w + (size_t)i * L_ * 3, masks);
        attn_apply_kernel<<<dim3(L_ / 16, B_ * H_), 32, 0, stream>>>(adjbf, vTbf, hbuf);
        ln_bf16_kernel<<<B_ * L_, DIM_, 0, stream>>>(hbuf, ln2_g + i * DIM_, ln2_b + i * DIM_, ybf);
        transpose_bf16_kernel<<<(DIM_ * DFF_ + 255) / 256, 256, 0, stream>>>(
            ffn1_w + (size_t)i * DIM_ * DFF_, w1T, DIM_, DFF_);
        transpose_bf16_kernel<<<(DFF_ * DIM_ + 255) / 256, 256, 0, stream>>>(
            ffn2_w + (size_t)i * DFF_ * DIM_, w2T, DFF_, DIM_);
        ffn1_wmma_kernel<<<dim3(B_ * L_ / 16, DFF_ / 16), 32, 0, stream>>>(ybf, w1T, ffn1_b + i * DFF_, midbf);
        ffn2_wmma_kernel<<<dim3(B_ * L_ / 16, DIM_ / 16), 32, 0, stream>>>(midbf, w2T, ffn2_b + i * DIM_, hbuf);
    }

    cast_bf16_kernel<<<(B_ * L_ * DIM_ + 255) / 256, 256, 0, stream>>>(hbuf, hbf, B_ * L_ * DIM_);
    transpose_bf16_kernel<<<(N_ * DIM_ * PRED_ + 255) / 256, 256, 0, stream>>>(
        head_w, hwT, N_ * DIM_, PRED_);
    head_wmma_kernel<<<dim3((B_ * C_) / 16, PRED_ / 16), 32, 0, stream>>>(
        hbf, hwT, head_b, meanb, stdb, out);
}
